// Qwen3MoE_42047729828451
// MI455X (gfx1250) — matrix operand order repaired and checked by reading, not compiled, not run on hardware
//
#include <hip/hip_runtime.h>
#include <hip/hip_bf16.h>
#include <stdint.h>

#define T_DIM 2048
#define H_DIM 2048
#define I_DIM 768
#define E_NUM 64
#define TOPK  8
#define TILE_M 32        // tokens per block: 2 WMMA row-blocks -> 2x weight reuse
#define SA (H_DIM + 4)   // LDS stride (elements) for A tile (pad: banks differ per row)
#define SH (I_DIM + 4)   // LDS stride (elements) for h tile

typedef __attribute__((ext_vector_type(16))) __bf16        v16bf;
typedef __attribute__((ext_vector_type(8)))  float         v8f;
typedef __attribute__((ext_vector_type(8)))  unsigned int  v8u;
typedef __attribute__((ext_vector_type(2)))  float         v2f;
typedef __attribute__((ext_vector_type(2)))  __bf16        v2bf;

// Packed f32x2 -> bf16x2 (single v_cvt_pk_bf16_f32, no half-register moves)
__device__ __forceinline__ unsigned pack2_bf16(float lo, float hi) {
    v2f f; f[0] = lo; f[1] = hi;
    v2bf b = __builtin_convertvector(f, v2bf);
    return __builtin_bit_cast(unsigned, b);
}
__device__ __forceinline__ unsigned short f2bf_bits(float f) {
    return __builtin_bit_cast(unsigned short, (__bf16)f);
}

// ---- WMMA fragment loaders (wave32; this code's contiguous k order, same in A and B, not the ISA table order) ----

// 16-bit A matrix 16x32: lane m = lane&15; lanes 0-15 hold K=0..15,
// lanes 16-31 hold K=16..31; VGPR r holds one K pair.
__device__ __forceinline__ v16bf load_a_frag(const unsigned short* S, int stride,
                                             int kbase, int lane) {
    const int m  = lane & 15;
    const int hi = lane >> 4;
    v8u u;
#pragma unroll
    for (int r = 0; r < 8; ++r) {
        int k = kbase + hi * 16 + r * 2;  // even => 4B aligned
        u[r] = *reinterpret_cast<const unsigned int*>(S + m * stride + k);
    }
    return __builtin_bit_cast(v16bf, u);
}

// 16-bit B matrix 32x16 (KxN): lane n = lane&15; lanes 0-15 hold K=0..15,
// lanes 16-31 hold K=16..31; VGPR r holds K pair (base+2r, base+2r+1).
// Source is fp32 row-major [K rows of length ldb]; convert to bf16 in-register.
__device__ __forceinline__ v16bf load_b_frag(const float* __restrict__ B, int ldb,
                                             int kbase, int n0, int lane) {
    const int n  = n0 + (lane & 15);
    const int kb = kbase + ((lane >> 4) ? 16 : 0);
    v8u u;
#pragma unroll
    for (int r = 0; r < 8; ++r) {
        u[r] = pack2_bf16(B[(size_t)(kb + 2 * r)     * ldb + n],
                          B[(size_t)(kb + 2 * r + 1) * ldb + n]);
    }
    return __builtin_bit_cast(v16bf, u);
}

__device__ __forceinline__ v8f wmma_bf16(v16bf a, v16bf b, v8f c) {
    return __builtin_amdgcn_wmma_f32_16x16x32_bf16(
        /*neg_a=*/false, a, /*neg_b=*/false, b,
        /*c_mod=*/(short)0, c, /*reuse_a=*/false, /*reuse_b=*/false);
}

// ----------------------------------------------------------------------------------
// Kernel 1: zero output accumulator + per-expert counters
__global__ void k_zero(float* __restrict__ out, int* __restrict__ counts, long n) {
    long i = (long)blockIdx.x * blockDim.x + threadIdx.x;
    if (i < n) out[i] = 0.0f;
    if (i < E_NUM) counts[i] = 0;
}

// Kernel 2: router logits = x @ gate_w   ([T,H] x [H,E] -> [T,E])
__global__ void k_router_logits(const float* __restrict__ x,
                                const float* __restrict__ gw,
                                float* __restrict__ logits) {
    const int t = blockIdx.x;          // one block per token
    const int e = threadIdx.x;         // blockDim.x == E_NUM
    const float* xr = x + (size_t)t * H_DIM;
    float acc = 0.0f;
#pragma unroll 4
    for (int h = 0; h < H_DIM; ++h)
        acc = fmaf(xr[h], gw[(size_t)h * E_NUM + e], acc);
    logits[(size_t)t * E_NUM + e] = acc;
}

// Kernel 3: one wave32 per token: top-8 via shfl_xor argmax reduction,
// softmax over selected, lane 0 scatters into per-expert lists.
__global__ void k_topk_scatter(const float* __restrict__ logits,
                               int* __restrict__ counts,
                               int* __restrict__ etok,
                               float* __restrict__ ewt) {
    const int lane = threadIdx.x & 31;
    const int t = blockIdx.x * (blockDim.x >> 5) + (threadIdx.x >> 5);
    if (t >= T_DIM) return;

    // Each lane owns 2 of the 64 expert logits
    float v0 = logits[(size_t)t * E_NUM + lane];
    float v1 = logits[(size_t)t * E_NUM + 32 + lane];

    float val[TOPK]; int idx[TOPK];
#pragma unroll
    for (int k = 0; k < TOPK; ++k) {
        float mv; int mi;
        if (v0 >= v1) { mv = v0; mi = lane; }         // lowest index wins ties
        else          { mv = v1; mi = lane + 32; }
#pragma unroll
        for (int off = 16; off > 0; off >>= 1) {
            float ov = __shfl_xor(mv, off, 32);
            int   oi = __shfl_xor(mi, off, 32);
            if (ov > mv || (ov == mv && oi < mi)) { mv = ov; mi = oi; }
        }
        val[k] = mv; idx[k] = mi;
        if (mi == lane)      v0 = -3.4e38f;
        if (mi == lane + 32) v1 = -3.4e38f;
    }
    float m = val[0], s = 0.0f, w[TOPK];
#pragma unroll
    for (int k = 0; k < TOPK; ++k) { w[k] = __expf(val[k] - m); s += w[k]; }
    float inv = __builtin_amdgcn_rcpf(s);
    if (lane == 0) {
#pragma unroll
        for (int k = 0; k < TOPK; ++k) {
            int e = idx[k];
            int pos = atomicAdd(&counts[e], 1);
            etok[(size_t)e * T_DIM + pos] = t;
            ewt [(size_t)e * T_DIM + pos] = w[k] * inv;
        }
    }
}

// Kernel 4: grouped expert SwiGLU FFN with bf16 WMMA, weighted scatter-add output.
// 32-token tile = two 16-row WMMA blocks sharing every B fragment (2x reuse).
__global__ __launch_bounds__(256)
void k_moe_ffn(const float* __restrict__ x,
               const float* __restrict__ w_gate,
               const float* __restrict__ w_up,
               const float* __restrict__ w_down,
               const int*   __restrict__ counts,
               const int*   __restrict__ etok,
               const float* __restrict__ ewt,
               float*       __restrict__ out) {
    __shared__ unsigned short As[TILE_M * SA];   // x tile, bf16, 32 x 2048 (+pad) ~128KB
    __shared__ unsigned short Hs[TILE_M * SH];   // h tile, bf16, 32 x 768  (+pad) ~48KB
    __shared__ int   Ts[TILE_M];
    __shared__ float Ws[TILE_M];

    const int e    = blockIdx.y;
    const int tile = blockIdx.x;
    const int cnt  = counts[e];
    if (tile * TILE_M >= cnt) return;            // uniform early-out (before barriers)

    const int tid  = threadIdx.x;
    const int lane = tid & 31;
    const int wv   = tid >> 5;                   // 8 wave32 per block
    const int nn   = lane & 15;
    const int hi   = lane >> 4;

    if (tid < TILE_M) {
        int r = tile * TILE_M + tid;
        if (r < cnt) {
            Ts[tid] = etok[(size_t)e * T_DIM + r];
            Ws[tid] = ewt [(size_t)e * T_DIM + r];
        } else {
            Ts[tid] = 0;                         // pad: valid row, zero combine weight
            Ws[tid] = 0.0f;
        }
    }
    __syncthreads();

    // Gather 32 token rows of x into LDS as bf16 (float4 loads, 8B LDS stores)
    for (int i = tid; i < TILE_M * (H_DIM / 4); i += 256) {
        int m  = i >> 9;                         // / (H_DIM/4) == 512
        int h4 = (i & 511) * 4;
        float4 f = *reinterpret_cast<const float4*>(x + (size_t)Ts[m] * H_DIM + h4);
        uint2 p;
        p.x = pack2_bf16(f.x, f.y);
        p.y = pack2_bf16(f.z, f.w);
        *reinterpret_cast<uint2*>(&As[m * SA + h4]) = p;
    }
    __syncthreads();

    const float* wg = w_gate + (size_t)e * H_DIM * I_DIM;
    const float* wu = w_up   + (size_t)e * H_DIM * I_DIM;
    const float* wd = w_down + (size_t)e * I_DIM * H_DIM;

    // ---- Phase 1: h = silu(x @ wg) * (x @ wu), 48 N-tiles over I, 6 per wave ----
    for (int ct = wv; ct < I_DIM / 16; ct += 8) {
        const int n0 = ct * 16;
        v8f accg0 = {}, accg1 = {}, accu0 = {}, accu1 = {};
#pragma unroll 2
        for (int kk = 0; kk < H_DIM; kk += 32) {
            if ((kk & 255) == 0) {               // prefetch weight stream ~512 rows ahead
                int pk = (kk + 512 < H_DIM) ? kk + 512 : kk;
                __builtin_prefetch(wg + (size_t)pk * I_DIM + n0, 0, 0);
                __builtin_prefetch(wu + (size_t)pk * I_DIM + n0, 0, 0);
            }
            v16bf a0 = load_a_frag(As,              SA, kk, lane);  // rows 0..15
            v16bf a1 = load_a_frag(As + 16 * SA,    SA, kk, lane);  // rows 16..31
            v16bf bg = load_b_frag(wg, I_DIM, kk, n0, lane);
            v16bf bu = load_b_frag(wu, I_DIM, kk, n0, lane);
            accg0 = wmma_bf16(a0, bg, accg0);
            accg1 = wmma_bf16(a1, bg, accg1);
            accu0 = wmma_bf16(a0, bu, accu0);
            accu1 = wmma_bf16(a1, bu, accu1);
        }
#pragma unroll
        for (int r = 0; r < 8; ++r) {
            int   m  = r + hi * 8;                              // C/D layout row
            float g0 = accg0[r];
            float h0 = g0 * __builtin_amdgcn_rcpf(1.0f + __expf(-g0)) * accu0[r];
            Hs[m * SH + n0 + nn] = f2bf_bits(h0);
            float g1 = accg1[r];
            float h1 = g1 * __builtin_amdgcn_rcpf(1.0f + __expf(-g1)) * accu1[r];
            Hs[(m + 16) * SH + n0 + nn] = f2bf_bits(h1);
        }
    }
    __syncthreads();

    // ---- Phase 2: out[tok] += w_tok * (h @ wd), 128 N-tiles over H, 16 per wave ----
    for (int ct = wv; ct < H_DIM / 16; ct += 8) {
        const int n0 = ct * 16;
        v8f acc0 = {}, acc1 = {};
#pragma unroll 2
        for (int kk = 0; kk < I_DIM; kk += 32) {
            if ((kk & 255) == 0) {
                int pk = (kk + 512 < I_DIM) ? kk + 512 : kk;
                __builtin_prefetch(wd + (size_t)pk * H_DIM + n0, 0, 0);
            }
            v16bf a0 = load_a_frag(Hs,           SH, kk, lane);
            v16bf a1 = load_a_frag(Hs + 16 * SH, SH, kk, lane);
            v16bf b  = load_b_frag(wd, H_DIM, kk, n0, lane);
            acc0 = wmma_bf16(a0, b, acc0);
            acc1 = wmma_bf16(a1, b, acc1);
        }
#pragma unroll
        for (int r = 0; r < 8; ++r) {
            int m = r + hi * 8;
            atomicAdd(&out[(size_t)Ts[m]      * H_DIM + n0 + nn], Ws[m]      * acc0[r]);
            atomicAdd(&out[(size_t)Ts[m + 16] * H_DIM + n0 + nn], Ws[m + 16] * acc1[r]);
        }
    }
}

// ----------------------------------------------------------------------------------
extern "C" void kernel_launch(void* const* d_in, const int* in_sizes, int n_in,
                              void* d_out, int out_size, void* d_ws, size_t ws_size,
                              hipStream_t stream) {
    const float* x      = (const float*)d_in[0];   // [T, H]
    const float* gate_w = (const float*)d_in[1];   // [H, E]
    const float* w_gate = (const float*)d_in[2];   // [E, H, I]
    const float* w_up   = (const float*)d_in[3];   // [E, H, I]
    const float* w_down = (const float*)d_in[4];   // [E, I, H]
    float* out = (float*)d_out;                    // [T, H]

    // Workspace layout (bytes): logits | counts | expert token lists | expert weights
    char*  ws     = (char*)d_ws;
    float* logits = (float*)ws;                                     // T*E f32 (512 KB)
    int*   counts = (int*)(ws + (size_t)T_DIM * E_NUM * 4);         // E i32
    int*   etok   = (int*)(ws + (size_t)T_DIM * E_NUM * 4 + 1024);  // E*T i32 (512 KB)
    float* ewt    = (float*)((char*)etok + (size_t)E_NUM * T_DIM * 4); // E*T f32

    const long n_out = (long)T_DIM * H_DIM;
    k_zero<<<(unsigned)((n_out + 255) / 256), 256, 0, stream>>>(out, counts, n_out);
    k_router_logits<<<T_DIM, E_NUM, 0, stream>>>(x, gate_w, logits);
    k_topk_scatter<<<(T_DIM + 127) / 128, 128, 0, stream>>>(logits, counts, etok, ewt);

    dim3 grid(T_DIM / TILE_M, E_NUM);   // 64 tile slots x 64 experts, early-out on count
    k_moe_ffn<<<grid, 256, 0, stream>>>(x, w_gate, w_up, w_down,
                                        counts, etok, ewt, out);
}
